// MultiFieldAttention_39788577030977
// MI455X (gfx1250) — matrix operand order repaired and checked by reading, not compiled, not run on hardware
//
#include <hip/hip_runtime.h>

typedef __attribute__((ext_vector_type(16))) _Float16 v16h;
typedef __attribute__((ext_vector_type(8)))  _Float16 v8h;
typedef __attribute__((ext_vector_type(8)))  float    v8f;

#define NNODE 12288
#define NCH   16
#define KNB   9
#define NV    16
#define ATTD  128
#define HD    32
#define NHEAD 4
#define NAW   4
#define NGRP  (NNODE / NAW)   // 3072
#define KVIN  144
#define KVINP 160             // zero-padded K for the kv GEMM
#define TOK   64

// packed B-fragment tile table (each tile = 512 f16 = 32 lanes x 16 elems)
#define TQ_BASE  0    // 8 tiles:  Wq cols nt*16, K=16 padded to 32
#define TKV_BASE 8    // 80 tiles: (part*8+nt)*5+ks, Wkv, K=144 padded to 160
#define TWO_BASE 88   // 4 tiles:  Wout k0=h*32, N=16
#define TW1_BASE 92   // 2 tiles:  W1 cols nt*16, K=16 padded to 32
#define TW2_BASE 94   // 1 tile:   W2, K=32, N=16
#define NTILES   95   // 48640 f16 total in d_ws

// ---- WMMA fragment helpers (wave32, A/B in contiguous k order k = 16*(lane/16) + elem) -------------

// A (16x32 f16), buf row-major [rows][ld], ld multiple of 32, k0 multiple of 32.
__device__ inline v16h load_A_f16(const _Float16* buf, int ld, int row0, int k0,
                                  int lane) {
  int r  = row0 + (lane & 15);
  int kb = (lane & 16) ? 16 : 0;
  const _Float16* p = buf + r * ld + k0 + kb;
  v8h lo = *(const v8h*)(p);
  v8h hi = *(const v8h*)(p + 8);
  return __builtin_shufflevector(lo, hi, 0, 1, 2, 3, 4, 5, 6, 7,
                                 8, 9, 10, 11, 12, 13, 14, 15);
}

// Pre-packed B fragment: one aligned 32-byte load per lane.
__device__ inline v16h load_B_pk(const _Float16* wt, int tile, int lane) {
  return *(const v16h*)(wt + tile * 512 + lane * 16);
}

// B[k][n] = S[n][k] with S row-major [N][ld]: contiguous 32-byte LDS load.
__device__ inline v16h load_B_kT(const _Float16* S, int ld, int k0, int n0,
                                 int lane) {
  int nn = n0 + (lane & 15);
  int kb = (lane & 16) ? 16 : 0;
  return *(const v16h*)(S + nn * ld + k0 + kb);
}

// C/D (16x16 f32): col N = lane&15, rows M = e + 8*(lane>=16)
__device__ inline void store_C_f16(_Float16* buf, int ld, int row0, int n0,
                                   v8f c, int lane) {
  int nn = n0 + (lane & 15);
  int rb = (lane & 16) ? 8 : 0;
#pragma unroll
  for (int e = 0; e < 8; ++e) buf[(row0 + rb + e) * ld + nn] = (_Float16)c[e];
}

// Transposed C store into [col][row] panel: one packed 16-byte store per lane.
__device__ inline void store_C_f16_T(_Float16* bufT, int ldT, int row0, int n0,
                                     v8f c, int lane) {
  int col = n0 + (lane & 15);
  int rb = (lane & 16) ? 8 : 0;
  v8h h;
#pragma unroll
  for (int e = 0; e < 8; ++e) h[e] = (_Float16)c[e];
  *(v8h*)(bufT + col * ldT + row0 + rb) = h;
}

__device__ inline v8f wmma16(v16h a, v16h b, v8f c) {
  return __builtin_amdgcn_wmma_f32_16x16x32_f16(false, a, false, b, (short)0, c,
                                                false, false);
}

// ---- weight f32 -> f16 pre-packed B fragments in workspace ----------------
__global__ void mfa_pack_weights(const float* __restrict__ Wq,
                                 const float* __restrict__ Wkv,
                                 const float* __restrict__ Wo,
                                 const float* __restrict__ W1,
                                 const float* __restrict__ W2,
                                 _Float16* __restrict__ ws) {
  int i = blockIdx.x * blockDim.x + threadIdx.x;
  if (i >= NTILES * 512) return;
  int tile = i >> 9;
  int rem  = i & 511;
  int lane = rem >> 4;
  int e    = rem & 15;
  int nn = lane & 15;
  int kb = (lane & 16) ? 16 : 0;
  const float* W; int ld, n0, k0, kr;
  if (tile < TKV_BASE)      { W = Wq;  ld = 128; n0 = tile * 16; k0 = 0; kr = 16; }
  else if (tile < TWO_BASE) { int t = tile - TKV_BASE; int pn = t / 5, ks = t % 5;
                              W = Wkv; ld = 256; n0 = pn * 16; k0 = ks * 32; kr = 144; }
  else if (tile < TW1_BASE) { W = Wo;  ld = 16;  n0 = 0; k0 = (tile - TWO_BASE) * 32; kr = 128; }
  else if (tile < TW2_BASE) { W = W1;  ld = 32;  n0 = (tile - TW1_BASE) * 16; k0 = 0; kr = 16; }
  else                      { W = W2;  ld = 16;  n0 = 0; k0 = 0; kr = 32; }
  int k = k0 + kb + e;
  ws[i] = (k < kr) ? (_Float16)W[k * ld + n0 + nn] : (_Float16)0.0f;
}

// ---- main kernel: one wave32 per attention group (4 nodes x 16 v = 64 tok) -
__global__ __launch_bounds__(32) void mfa_main(
    const float* __restrict__ x, const int* __restrict__ adjc,
    const float* __restrict__ ln1_s, const float* __restrict__ ln1_b,
    const float* __restrict__ b_kv, const float* __restrict__ b_out,
    const float* __restrict__ gamma, const float* __restrict__ ln2_s,
    const float* __restrict__ ln2_b, const float* __restrict__ b1,
    const float* __restrict__ b2, const float* __restrict__ gamma_mlp,
    const _Float16* __restrict__ wf, float* __restrict__ out) {
  __shared__ __align__(16) char smem[59392];
  _Float16* qn   = (_Float16*)(smem);            // [64][32]  LN1 rows, cols16-31 = 0
  _Float16* kvb  = (_Float16*)(smem + 4096);     // [64][160] gathered, cols144-159 = 0
  float*    atto = (float*)(smem + 4096);        // [64][16]  overlay after kvb freed
  float*    xob  = (float*)(smem + 8192);        // [64][16]  overlay
  _Float16* hb   = (_Float16*)(smem + 12288);    // [64][32]  overlay, cols16-31 = 0
  _Float16* mb   = (_Float16*)(smem + 16384);    // [64][32]  overlay
  _Float16* kb   = (_Float16*)(smem + 24576);    // [64][128] K panel [token][dim]
  _Float16* vbT  = (_Float16*)(smem + 40960);    // [128][64] V panel [dim][token]
  _Float16* scr  = (_Float16*)(smem + 57344);    // [16][64] tile scratch

  const int lane = threadIdx.x;
  const int g = blockIdx.x;
  const v8h zero8 = {};

  // hoist LN1 params (uniform loads, once)
  float l1s[NCH], l1b[NCH];
#pragma unroll
  for (int q4 = 0; q4 < 4; ++q4) {
    float4 fs = ((const float4*)ln1_s)[q4];
    float4 fb = ((const float4*)ln1_b)[q4];
    l1s[q4 * 4 + 0] = fs.x; l1s[q4 * 4 + 1] = fs.y;
    l1s[q4 * 4 + 2] = fs.z; l1s[q4 * 4 + 3] = fs.w;
    l1b[q4 * 4 + 0] = fb.x; l1b[q4 * 4 + 1] = fb.y;
    l1b[q4 * 4 + 2] = fb.z; l1b[q4 * 4 + 3] = fb.w;
  }

  // phase 1: LN1 of the 64 group token rows -> qn (f16, K-padded)
  for (int r = lane; r < TOK; r += 32) {
    int v = r >> 2, a = r & 3, node = g * NAW + a;
    const float4* xr = (const float4*)(x + ((size_t)v * NNODE + node) * NCH);
    float val[NCH]; float s = 0.f, s2 = 0.f;
#pragma unroll
    for (int q4 = 0; q4 < 4; ++q4) {
      float4 f = xr[q4];
      val[q4 * 4 + 0] = f.x; val[q4 * 4 + 1] = f.y;
      val[q4 * 4 + 2] = f.z; val[q4 * 4 + 3] = f.w;
    }
#pragma unroll
    for (int c = 0; c < NCH; ++c) { s += val[c]; s2 += val[c] * val[c]; }
    float mean = s * (1.f / NCH);
    float var  = s2 * (1.f / NCH) - mean * mean;
    float rstd = rsqrtf(var + 1e-5f);
#pragma unroll
    for (int c = 0; c < NCH; ++c)
      qn[r * 32 + c] = (_Float16)((val[c] - mean) * rstd * l1s[c] + l1b[c]);
    *(v8h*)(qn + r * 32 + 16) = zero8;
    *(v8h*)(qn + r * 32 + 24) = zero8;
    // zero the kvb K-pad tail for this token row
    *(v8h*)(kvb + r * KVINP + 144) = zero8;
    *(v8h*)(kvb + r * KVINP + 152) = zero8;
  }
  // phase 2: gather K=9 neighbors, LN1 them -> kvb
  for (int idx = lane; idx < TOK * KNB; idx += 32) {
    int t = idx / KNB, kn = idx % KNB;
    int v = t >> 2, a = t & 3;
    int nb = adjc[(g * NAW + a) * KNB + kn];
    const float4* xr = (const float4*)(x + ((size_t)v * NNODE + nb) * NCH);
    float val[NCH]; float s = 0.f, s2 = 0.f;
#pragma unroll
    for (int q4 = 0; q4 < 4; ++q4) {
      float4 f = xr[q4];
      val[q4 * 4 + 0] = f.x; val[q4 * 4 + 1] = f.y;
      val[q4 * 4 + 2] = f.z; val[q4 * 4 + 3] = f.w;
    }
#pragma unroll
    for (int c = 0; c < NCH; ++c) { s += val[c]; s2 += val[c] * val[c]; }
    float mean = s * (1.f / NCH);
    float var  = s2 * (1.f / NCH) - mean * mean;
    float rstd = rsqrtf(var + 1e-5f);
#pragma unroll
    for (int c = 0; c < NCH; ++c)
      kvb[t * KVINP + kn * NCH + c] =
          (_Float16)((val[c] - mean) * rstd * l1s[c] + l1b[c]);
  }
  __syncthreads();

  // phase 3a: K = kvb(64x160z) @ Wkv[:, :128] + b_kv -> kb [token][dim]
  for (int nt = 0; nt < 8; ++nt) {
    for (int mt = 0; mt < 4; ++mt) {
      v8f acc = {};
      for (int ks = 0; ks < 5; ++ks) {
        v16h a = load_A_f16(kvb, KVINP, mt * 16, ks * 32, lane);
        acc = wmma16(a, load_B_pk(wf, TKV_BASE + nt * 5 + ks, lane), acc);
      }
      float bv = b_kv[nt * 16 + (lane & 15)];
#pragma unroll
      for (int e = 0; e < 8; ++e) acc[e] += bv;
      store_C_f16(kb, ATTD, mt * 16, nt * 16, acc, lane);
    }
  }
  // phase 3b: V = kvb @ Wkv[:, 128:] + b_kv -> vbT [dim][token] (transposed)
  for (int nt = 0; nt < 8; ++nt) {
    for (int mt = 0; mt < 4; ++mt) {
      v8f acc = {};
      for (int ks = 0; ks < 5; ++ks) {
        v16h a = load_A_f16(kvb, KVINP, mt * 16, ks * 32, lane);
        acc = wmma16(a, load_B_pk(wf, TKV_BASE + (8 + nt) * 5 + ks, lane), acc);
      }
      float bv = b_kv[128 + nt * 16 + (lane & 15)];
#pragma unroll
      for (int e = 0; e < 8; ++e) acc[e] += bv;
      store_C_f16_T(vbT, TOK, mt * 16, nt * 16, acc, lane);
    }
  }
  __syncthreads();

  // phase 4: attention (per row-tile mt, per head h) + out-projection
  const float scl = 0.17677669529663687f;  // 1/sqrt(32)
  for (int mt = 0; mt < 4; ++mt) {
    v8f oacc = {};  // att_out tile (16x16), accumulated over heads (K=128)
    for (int h = 0; h < NHEAD; ++h) {
      // q-projection subtile 16x32 on the fly: qn(16x32z) @ Wq[:, h*32..]
      v16h aq = load_A_f16(qn, 32, mt * 16, 0, lane);
      for (int t2 = 0; t2 < 2; ++t2) {
        v8f c = {};
        c = wmma16(aq, load_B_pk(wf, TQ_BASE + h * 2 + t2, lane), c);
        store_C_f16(scr, 32, 0, t2 * 16, c, lane);
      }
      __syncthreads();
      // scores tile 16x64 = qp(16x32) @ K_h^T(32x64)
      v16h aS = load_A_f16(scr, 32, 0, 0, lane);
      v8f sc0 = {}, sc1 = {}, sc2 = {}, sc3 = {};
      sc0 = wmma16(aS, load_B_kT(kb, ATTD, h * HD, 0,  lane), sc0);
      sc1 = wmma16(aS, load_B_kT(kb, ATTD, h * HD, 16, lane), sc1);
      sc2 = wmma16(aS, load_B_kT(kb, ATTD, h * HD, 32, lane), sc2);
      sc3 = wmma16(aS, load_B_kT(kb, ATTD, h * HD, 48, lane), sc3);
      __syncthreads();
      // softmax along 64 keys: row M = e + 8*half lives in one 16-lane half
      int rb = (lane & 16) ? 8 : 0;
      int nn = lane & 15;
#pragma unroll
      for (int e = 0; e < 8; ++e) {
        float x0 = sc0[e] * scl, x1 = sc1[e] * scl, x2 = sc2[e] * scl, x3 = sc3[e] * scl;
        float mx = fmaxf(fmaxf(x0, x1), fmaxf(x2, x3));
#pragma unroll
        for (int o = 1; o < 16; o <<= 1) mx = fmaxf(mx, __shfl_xor(mx, o, 32));
        float e0 = __expf(x0 - mx), e1 = __expf(x1 - mx);
        float e2 = __expf(x2 - mx), e3 = __expf(x3 - mx);
        float sm = e0 + e1 + e2 + e3;
#pragma unroll
        for (int o = 1; o < 16; o <<= 1) sm += __shfl_xor(sm, o, 32);
        float inv = 1.f / sm;
        int row = rb + e;
        scr[row * 64 + nn]      = (_Float16)(e0 * inv);
        scr[row * 64 + 16 + nn] = (_Float16)(e1 * inv);
        scr[row * 64 + 32 + nn] = (_Float16)(e2 * inv);
        scr[row * 64 + 48 + nn] = (_Float16)(e3 * inv);
      }
      __syncthreads();
      // out tile 16x32 = att(16x64) @ V_h(64x32)  (V from transposed panel)
      v8f ov0 = {}, ov1 = {};
      for (int ks = 0; ks < 2; ++ks) {
        v16h aA = load_A_f16(scr, 64, 0, ks * 32, lane);
        ov0 = wmma16(aA, load_B_kT(vbT, TOK, ks * 32, h * HD + 0,  lane), ov0);
        ov1 = wmma16(aA, load_B_kT(vbT, TOK, ks * 32, h * HD + 16, lane), ov1);
      }
      __syncthreads();
      store_C_f16(scr, 32, 0, 0,  ov0, lane);
      store_C_f16(scr, 32, 0, 16, ov1, lane);
      __syncthreads();
      // accumulate att_out += out_h(16x32) @ W_out[h*32.., :16]
      v16h aO = load_A_f16(scr, 32, 0, 0, lane);
      oacc = wmma16(aO, load_B_pk(wf, TWO_BASE + h, lane), oacc);
      __syncthreads();
    }
    {
      int nn = lane & 15;
      int rb = (lane & 16) ? 8 : 0;
      float bo = b_out[nn];
#pragma unroll
      for (int e = 0; e < 8; ++e)
        atto[(mt * 16 + rb + e) * NCH + nn] = oacc[e] + bo;
    }
  }
  __syncthreads();

  // phase 5: gamma residual + LN2 -> xob (f32), hb (f16, K-padded)
  float gmm[NCH], l2s[NCH], l2b[NCH];
#pragma unroll
  for (int q4 = 0; q4 < 4; ++q4) {
    float4 fg = ((const float4*)gamma)[q4];
    float4 fs = ((const float4*)ln2_s)[q4];
    float4 fb = ((const float4*)ln2_b)[q4];
    gmm[q4 * 4 + 0] = fg.x; gmm[q4 * 4 + 1] = fg.y;
    gmm[q4 * 4 + 2] = fg.z; gmm[q4 * 4 + 3] = fg.w;
    l2s[q4 * 4 + 0] = fs.x; l2s[q4 * 4 + 1] = fs.y;
    l2s[q4 * 4 + 2] = fs.z; l2s[q4 * 4 + 3] = fs.w;
    l2b[q4 * 4 + 0] = fb.x; l2b[q4 * 4 + 1] = fb.y;
    l2b[q4 * 4 + 2] = fb.z; l2b[q4 * 4 + 3] = fb.w;
  }
  for (int r = lane; r < TOK; r += 32) {
    int v = r >> 2, a = r & 3, node = g * NAW + a;
    const float4* xr = (const float4*)(x + ((size_t)v * NNODE + node) * NCH);
    float xo[NCH]; float s = 0.f, s2 = 0.f;
#pragma unroll
    for (int q4 = 0; q4 < 4; ++q4) {
      float4 f = xr[q4];
      xo[q4 * 4 + 0] = f.x; xo[q4 * 4 + 1] = f.y;
      xo[q4 * 4 + 2] = f.z; xo[q4 * 4 + 3] = f.w;
    }
#pragma unroll
    for (int c = 0; c < NCH; ++c) {
      float t = xo[c] + gmm[c] * atto[r * NCH + c];
      xo[c] = t; s += t; s2 += t * t;
    }
    float mean = s * (1.f / NCH);
    float var  = s2 * (1.f / NCH) - mean * mean;
    float rstd = rsqrtf(var + 1e-5f);
#pragma unroll
    for (int c = 0; c < NCH; ++c) {
      xob[r * NCH + c] = xo[c];
      hb[r * 32 + c] = (_Float16)((xo[c] - mean) * rstd * l2s[c] + l2b[c]);
    }
    *(v8h*)(hb + r * 32 + 16) = zero8;
    *(v8h*)(hb + r * 32 + 24) = zero8;
  }
  __syncthreads();

  // phase 6: MLP via WMMA: h(64x32z) @ W1 -> gelu -> @ W2
  for (int mt = 0; mt < 4; ++mt) {
    v16h aH = load_A_f16(hb, 32, mt * 16, 0, lane);
    for (int nt = 0; nt < 2; ++nt) {
      v8f c = {};
      c = wmma16(aH, load_B_pk(wf, TW1_BASE + nt, lane), c);
      float bb = b1[nt * 16 + (lane & 15)];
#pragma unroll
      for (int e = 0; e < 8; ++e) {
        float t = c[e] + bb;
        float t3 = t * t * t;  // tanh-approx gelu (jax default)
        c[e] = 0.5f * t * (1.f + tanhf(0.7978845608028654f * (t + 0.044715f * t3)));
      }
      store_C_f16(mb, 32, mt * 16, nt * 16, c, lane);
    }
  }
  __syncthreads();
  {
    int nn = lane & 15;
    int rb = (lane & 16) ? 8 : 0;
    float b2c = b2[nn], gmc = gamma_mlp[nn];
    for (int mt = 0; mt < 4; ++mt) {
      v16h aM = load_A_f16(mb, 32, mt * 16, 0, lane);
      v8f c = {};
      c = wmma16(aM, load_B_pk(wf, TW2_BASE, lane), c);
#pragma unroll
      for (int e = 0; e < 8; ++e) {
        int r = mt * 16 + rb + e;
        int v = r >> 2, a = r & 3, node = g * NAW + a;
        out[((size_t)v * NNODE + node) * NCH + nn] =
            xob[r * NCH + nn] + gmc * (c[e] + b2c);
      }
    }
  }
}

extern "C" void kernel_launch(void* const* d_in, const int* in_sizes, int n_in,
                              void* d_out, int out_size, void* d_ws, size_t ws_size,
                              hipStream_t stream) {
  (void)in_sizes; (void)n_in; (void)out_size; (void)ws_size;
  const float* x     = (const float*)d_in[0];
  const int*   adjc  = (const int*)d_in[1];
  const float* ln1_s = (const float*)d_in[2];
  const float* ln1_b = (const float*)d_in[3];
  const float* Wq    = (const float*)d_in[4];
  const float* Wkv   = (const float*)d_in[5];
  const float* bkv   = (const float*)d_in[6];
  const float* Wo    = (const float*)d_in[7];
  const float* bo    = (const float*)d_in[8];
  const float* gam   = (const float*)d_in[9];
  const float* ln2_s = (const float*)d_in[10];
  const float* ln2_b = (const float*)d_in[11];
  const float* W1    = (const float*)d_in[12];
  const float* b1    = (const float*)d_in[13];
  const float* W2    = (const float*)d_in[14];
  const float* b2    = (const float*)d_in[15];
  const float* gmlp  = (const float*)d_in[16];
  _Float16* wf = (_Float16*)d_ws;

  mfa_pack_weights<<<(NTILES * 512 + 255) / 256, 256, 0, stream>>>(Wq, Wkv, Wo,
                                                                   W1, W2, wf);
  mfa_main<<<NGRP, 32, 0, stream>>>(x, adjc, ln1_s, ln1_b, bkv, bo, gam, ln2_s,
                                    ln2_b, b1, b2, gmlp, wf, (float*)d_out);
}
